// NeuralNetwork_5712306504081
// MI455X (gfx1250) — matrix operand order repaired and checked by reading, not compiled, not run on hardware
//
#include <hip/hip_runtime.h>
#include <math.h>

#define Bsz 512
#define Ssz 128
#define Hsz 1024
#define Lsz 20
#define Osz 6

typedef __attribute__((ext_vector_type(16))) __bf16 v16bf;
typedef __attribute__((ext_vector_type(8)))  float  v8f;

__constant__ float c_sector[4] = {0.92f, 1.08f, 0.98f, 1.05f};

// ---------- bf16 helpers (bit-exact RNE, no reliance on __bf16 conversions) ----
static __device__ __forceinline__ __bf16 f2bf(float f) {
  union { float f; unsigned u; } c; c.f = f;
  unsigned r = c.u + 0x7fffu + ((c.u >> 16) & 1u);
  union { unsigned short s; __bf16 b; } o; o.s = (unsigned short)(r >> 16);
  return o.b;
}
static __device__ __forceinline__ float bf2f(__bf16 b) {
  union { __bf16 b; unsigned short s; } i; i.b = b;
  union { unsigned u; float f; } o; o.u = ((unsigned)i.s) << 16;
  return o.f;
}
static __device__ __forceinline__ void split2(float v, __bf16& hi, __bf16& lo) {
  hi = f2bf(v);
  lo = f2bf(v - bf2f(hi));
}

union Frag { v16bf v; uint4 q[2]; };
union TPack { uint4 q; __bf16 e[8]; };

// =============================================================================
// LDS-staged block GEMM mainloop (bf16x3 split-precision, f32 accumulate).
// Block = 256 threads = 8 waves in a 2x4 tile grid -> 32x64 macro-tile.
// A operand: hi panel at A, lo panel at A + AOFS (elements), rows = M dim,
//            K contiguous with leading dim LDA.
// B operand: hi at B, lo at B + BOFS, rows = N dim (B(k,n)=Bp[n][k]), ld LDB.
// Per K=32 step: stage A(32x32) + B(64x32) hi/lo panels into ping-pong LDS,
// one barrier per step; next step's global loads overlap the WMMAs.
// All hi/lo and buffer selects are integer offsets on a single base pointer
// so every memory op stays GLOBAL / DS (no flat fallback).
// LDS rows padded to 40 elements (80B) -> conflict-free ds_load_b128.
// =============================================================================
template<int KTOT, int LDA, int LDB, int AOFS, int BOFS>
static __device__ __forceinline__ v8f wmma_block_gemm(
    const __bf16* __restrict__ A, const __bf16* __restrict__ B) {
  __shared__ __bf16 sA[2][2 * 32 * 40];   // [buf][hi(0)/lo(1280) + row*40 + k]
  __shared__ __bf16 sB[2][2 * 64 * 40];   // [buf][hi(0)/lo(2560) + row*40 + k]

  const int tid  = threadIdx.x;
  const int lane = tid & 31;
  const int wv   = tid >> 5;
  const int lid  = lane & 15;
  const int half = lane >> 4;
  const int wr   = wv >> 2;   // 0..1  (m tile within macro-tile)
  const int wc   = wv & 3;    // 0..3  (n tile within macro-tile)

  // --- staged-load chunk decode (3 uint4 per thread per K-step) ---
  // j=0: A hi (tid<128) / A lo (tid>=128);  j=1: B hi;  j=2: B lo.
  const int accA  = tid & 127;
  const int arowc = accA >> 2, asegc = accA & 3;
  const int aLo   = (tid >> 7) & 1;                 // 0 = hi, 1 = lo
  const __bf16* gpA = A + (size_t)(aLo ? AOFS : 0) + (size_t)arowc * LDA + asegc * 8;
  const int     laA = aLo * 1280 + arowc * 40 + asegc * 8;

  const int browc = tid >> 2, bsegc = tid & 3;
  const __bf16* gpBh = B + (size_t)browc * LDB + bsegc * 8;
  const __bf16* gpBl = B + (size_t)BOFS + (size_t)browc * LDB + bsegc * 8;
  const int     laBh = browc * 40 + bsegc * 8;
  const int     laBl = 2560 + browc * 40 + bsegc * 8;

  uint4 stA  = *(const uint4*)(gpA);
  uint4 stBh = *(const uint4*)(gpBh);
  uint4 stBl = *(const uint4*)(gpBl);

  const int arow = (wr * 16 + lid) * 40;
  const int brow = (wc * 16 + lid) * 40;
  const int ka   = half * 8;
  const int kb   = half * 8;

  v8f acc = {};
  const int NS = KTOT / 32;
#pragma unroll 2
  for (int i = 0; i < NS; ++i) {
    const int buf = i & 1;
    // commit staged regs to this step's LDS buffer
    *(uint4*)&sA[buf][laA]  = stA;
    *(uint4*)&sB[buf][laBh] = stBh;
    *(uint4*)&sB[buf][laBl] = stBl;
    __syncthreads();
    // issue next step's global loads (overlap with ds reads + wmma below)
    if (i + 1 < NS) {
      stA  = *(const uint4*)(gpA  + (i + 1) * 32);
      stBh = *(const uint4*)(gpBh + (i + 1) * 32);
      stBl = *(const uint4*)(gpBl + (i + 1) * 32);
      __builtin_prefetch(gpA  + (i + 2) * 32, 0, 1);
      __builtin_prefetch(gpBh + (i + 2) * 32, 0, 1);
      __builtin_prefetch(gpBl + (i + 2) * 32, 0, 1);
    }
    // fragments from LDS
    Frag fah, fal, fbh, fbl;
    fah.q[0] = *(const uint4*)&sA[buf][arow + ka];
    fah.q[1] = *(const uint4*)&sA[buf][arow + ka + 16];
    fal.q[0] = *(const uint4*)&sA[buf][1280 + arow + ka];
    fal.q[1] = *(const uint4*)&sA[buf][1280 + arow + ka + 16];
    fbh.q[0] = *(const uint4*)&sB[buf][brow + kb];
    fbh.q[1] = *(const uint4*)&sB[buf][brow + kb + 16];
    fbl.q[0] = *(const uint4*)&sB[buf][2560 + brow + kb];
    fbl.q[1] = *(const uint4*)&sB[buf][2560 + brow + kb + 16];
    acc = __builtin_amdgcn_wmma_f32_16x16x32_bf16(false, fah.v, false, fbh.v, (short)0, acc, false, false);
    acc = __builtin_amdgcn_wmma_f32_16x16x32_bf16(false, fah.v, false, fbl.v, (short)0, acc, false, false);
    acc = __builtin_amdgcn_wmma_f32_16x16x32_bf16(false, fal.v, false, fbh.v, (short)0, acc, false, false);
  }
  return acc;
}

// ---------- embedding gather + mean pool, produce x (f32) + bf16 splits -------
__global__ void k_embed(const int* __restrict__ X, const float* __restrict__ emb,
                        float* __restrict__ x,
                        __bf16* __restrict__ xh, __bf16* __restrict__ xl,
                        __bf16* __restrict__ xTh, __bf16* __restrict__ xTl) {
  int b = blockIdx.x;
  int h = blockIdx.y * blockDim.x + threadIdx.x;
  const int* row = X + b * Ssz;
  float acc = 0.f;
#pragma unroll 4
  for (int s = 0; s < Ssz; ++s) {
    int tok = row[s];                       // uniform per block -> scalar load
    acc += emb[(size_t)tok * Hsz + h];
  }
  float v = acc * (1.0f / (float)Ssz);
  x[(size_t)b * Hsz + h] = v;
  __bf16 hi, lo; split2(v, hi, lo);
  xh[(size_t)b * Hsz + h] = hi;  xl[(size_t)b * Hsz + h] = lo;
  xTh[(size_t)h * Bsz + b] = hi; xTl[(size_t)h * Bsz + b] = lo;
}

// ---------- we = wi * sin(pi * x * y * wh), stored TRANSPOSED [H][B] ----------
__global__ void k_we(const float* __restrict__ x, const int* __restrict__ emo,
                     const float* __restrict__ wave_i, const float* __restrict__ wave_h,
                     int layer, __bf16* __restrict__ weTh, __bf16* __restrict__ weTl) {
  int idx = blockIdx.x * blockDim.x + threadIdx.x;
  int b = idx / Hsz, i = idx % Hsz;
  float wi = wave_i[layer], whc = wave_h[layer];
  float y = (float)emo[b];
  float v = wi * sinf(3.14159265358979f * x[idx] * y * whc);
  __bf16 hi, lo; split2(v, hi, lo);
  weTh[(size_t)i * Bsz + b] = hi;
  weTl[(size_t)i * Bsz + b] = lo;
}

// ---------- b_eff[i] = wave_b[l,i] + mean_b we[b,i]  (f32 recompute) ----------
__global__ void k_db(const float* __restrict__ x, const int* __restrict__ emo,
                     const float* __restrict__ wave_i, const float* __restrict__ wave_h,
                     const float* __restrict__ wave_b, int layer,
                     float* __restrict__ beff) {
  int i = blockIdx.x * blockDim.x + threadIdx.x;
  float wi = wave_i[layer], whc = wave_h[layer];
  float s = 0.f;
  for (int b = 0; b < Bsz; ++b) {
    float y = (float)emo[b];
    s += wi * sinf(3.14159265358979f * x[(size_t)b * Hsz + i] * y * whc);
  }
  beff[i] = wave_b[(size_t)layer * Hsz + i] + s * (1.0f / (float)Bsz);
}

// ---------- GEMM 1: W_eff = wave_W[l] + (weT . x)/B, split to bf16 ------------
// M=H (rows of weT), N=H, K=Bsz. A=[m][k]=weT, B(k,n)=xT[n][k].
// weT lo at +Hsz*Bsz, xT lo at +Hsz*Bsz (contiguous hi|lo allocations).
__global__ void k_gemm_dw(const __bf16* __restrict__ weT, const __bf16* __restrict__ xT,
                          const float* __restrict__ waveW, int layer,
                          __bf16* __restrict__ Wh, __bf16* __restrict__ Wl) {
  int bm = blockIdx.x & 31;       // 32 macro-rows of 32
  int bn = blockIdx.x >> 5;       // 16 macro-cols of 64
  v8f acc = wmma_block_gemm<Bsz, Bsz, Bsz, Hsz * Bsz, Hsz * Bsz>(
      weT + (size_t)(bm * 32) * Bsz,
      xT  + (size_t)(bn * 64) * Bsz);

  int lane = threadIdx.x & 31, wv = threadIdx.x >> 5;
  int lid = lane & 15, half = lane >> 4;
  int m0 = bm * 32 + (wv >> 2) * 16;
  int n0 = bn * 64 + (wv & 3) * 16;

  const float invB = 1.0f / (float)Bsz;
  int n = n0 + lid;
  const float* wrow = waveW + (size_t)layer * Hsz * Hsz;
#pragma unroll
  for (int r = 0; r < 8; ++r) {
    int m = m0 + half * 8 + r;
    float w = wrow[(size_t)m * Hsz + n] + acc[r] * invB;
    __bf16 hi, lo; split2(w, hi, lo);
    Wh[(size_t)m * Hsz + n] = hi;
    Wl[(size_t)m * Hsz + n] = lo;
  }
}

// ---------- GEMM 2: x_new = act(x . W_eff^T + b_eff) with PReLU --------------
// M=Bsz, N=H, K=H. A=[b][k] x splits (lo at +Bsz*Hsz), B(k,n)=W[n][k]
// (row-major [out][in], lo at +Hsz*Hsz).
__global__ void k_layer_fwd(const __bf16* __restrict__ xsp, const __bf16* __restrict__ W,
                            const float* __restrict__ beff,
                            const int* __restrict__ emo, const int* __restrict__ lab,
                            const float* __restrict__ a_table, int layer, int fsel,
                            float* __restrict__ x_out,
                            __bf16* __restrict__ xh_o, __bf16* __restrict__ xl_o,
                            __bf16* __restrict__ xTh_o, __bf16* __restrict__ xTl_o) {
  int bm = blockIdx.x & 15;       // 16 macro-rows of 32 (M=512)
  int bn = blockIdx.x >> 4;       // 16 macro-cols of 64 (N=1024)
  v8f acc = wmma_block_gemm<Hsz, Hsz, Hsz, Bsz * Hsz, Hsz * Hsz>(
      xsp + (size_t)(bm * 32) * Hsz,
      W   + (size_t)(bn * 64) * Hsz);

  int lane = threadIdx.x & 31, wv = threadIdx.x >> 5;
  int lid = lane & 15, half = lane >> 4;
  int m0 = bm * 32 + (wv >> 2) * 16;
  int n0 = bn * 64 + (wv & 3) * 16;

  int n = n0 + lid;
  float bias = beff[n];
  TPack th, tl;
#pragma unroll
  for (int r = 0; r < 8; ++r) {
    int m = m0 + half * 8 + r;
    float v = acc[r] + bias;
    float z;
    if (fsel == 0)      z = tanhf(v);
    else if (fsel == 1) z = sinf(v);
    else                z = fmaxf(v, 0.f);
    float slope = a_table[emo[m] * Lsz + layer] * c_sector[lab[m]];
    float o = (z > 0.f) ? z : slope * z;
    x_out[(size_t)m * Hsz + n] = o;
    __bf16 hi, lo; split2(o, hi, lo);
    xh_o[(size_t)m * Hsz + n] = hi;
    xl_o[(size_t)m * Hsz + n] = lo;
    th.e[r] = hi; tl.e[r] = lo;
  }
  // transposed splits: per lane 8 consecutive m -> one b128 store each
  *(uint4*)(xTh_o + (size_t)n * Bsz + m0 + half * 8) = th.q;
  *(uint4*)(xTl_o + (size_t)n * Bsz + m0 + half * 8) = tl.q;
}

// ---------- tiny final FC: out[b,o] = x[b,:] . fc_W[o,:] + fc_b[o] -----------
__global__ void k_fc(const float* __restrict__ x, const float* __restrict__ fcW,
                     const float* __restrict__ fcb, float* __restrict__ out) {
  int idx = blockIdx.x * blockDim.x + threadIdx.x;
  if (idx >= Bsz * Osz) return;
  int b = idx / Osz, o = idx % Osz;
  float s = fcb[o];
#pragma unroll 4
  for (int h = 0; h < Hsz; ++h)
    s += x[(size_t)b * Hsz + h] * fcW[(size_t)o * Hsz + h];
  out[idx] = s;
}

// -----------------------------------------------------------------------------
extern "C" void kernel_launch(void* const* d_in, const int* in_sizes, int n_in,
                              void* d_out, int out_size, void* d_ws, size_t ws_size,
                              hipStream_t stream) {
  const int*   X        = (const int*)d_in[0];
  const int*   emo_ids  = (const int*)d_in[1];
  const int*   emo_lab  = (const int*)d_in[2];
  const float* embed    = (const float*)d_in[3];
  const float* wave_W   = (const float*)d_in[4];
  const float* wave_b   = (const float*)d_in[5];
  const float* wave_i   = (const float*)d_in[6];
  const float* wave_h   = (const float*)d_in[7];
  const float* a_table  = (const float*)d_in[8];
  const float* fc_W     = (const float*)d_in[9];
  const float* fc_b     = (const float*)d_in[10];

  char* w = (char*)d_ws;
  size_t off = 0;
  auto take = [&](size_t bytes) -> void* {
    off = (off + 255) & ~(size_t)255;
    void* p = w + off;
    off += bytes;
    return p;
  };
  const size_t NBH = (size_t)Bsz * Hsz;   // elements per [B,H] / [H,B] panel
  const size_t NHH = (size_t)Hsz * Hsz;

  float*  x    = (float*)take(NBH * 4);
  // hi|lo pairs allocated CONTIGUOUSLY: lo = hi + N elements (sizes are
  // multiples of 256B so take() introduces no padding between them).
  __bf16* xs0  = (__bf16*)take(2 * NBH * 2);   // xh0 | xl0
  __bf16* xs1  = (__bf16*)take(2 * NBH * 2);   // xh1 | xl1
  __bf16* xT   = (__bf16*)take(2 * NBH * 2);   // xTh | xTl
  __bf16* weT  = (__bf16*)take(2 * NBH * 2);   // weTh | weTl
  __bf16* W    = (__bf16*)take(2 * NHH * 2);   // Wh | Wl
  float*  beff = (float*)take((size_t)Hsz * 4);

  __bf16* xsp[2] = { xs0, xs1 };

  k_embed<<<dim3(Bsz, Hsz / 256), 256, 0, stream>>>(X, embed, x,
                                                    xs0, xs0 + NBH, xT, xT + NBH);

  for (int l = 0; l < Lsz; ++l) {
    int cur = l & 1, nxt = cur ^ 1;
    k_we<<<(Bsz * Hsz) / 256, 256, 0, stream>>>(x, emo_ids, wave_i, wave_h, l,
                                                weT, weT + NBH);
    k_db<<<Hsz / 256, 256, 0, stream>>>(x, emo_ids, wave_i, wave_h, wave_b, l, beff);
    k_gemm_dw<<<32 * 16, 256, 0, stream>>>(weT, xT, wave_W, l, W, W + NHH);
    k_layer_fwd<<<16 * 16, 256, 0, stream>>>(xsp[cur], W, beff,
                                             emo_ids, emo_lab, a_table, l, l % 3,
                                             x, xsp[nxt], xsp[nxt] + NBH,
                                             xT, xT + NBH);
  }

  k_fc<<<(Bsz * Osz + 255) / 256, 256, 0, stream>>>(x, fc_W, fc_b, (float*)d_out);
}
